// FrostQuantixAttentionJAX_84129819394098
// MI455X (gfx1250) — matrix operand order repaired and checked by reading, not compiled, not run on hardware
//
#include <hip/hip_runtime.h>
#include <hip/hip_bf16.h>
#include <math.h>

// Sizes fixed by the reference: B=8, T=1024, C=768, H=12, d=64
#define NHEAD 12
#define TDIM  1024
#define CDIM  768
#define DHEAD 64

typedef __bf16 bf16;
typedef __attribute__((ext_vector_type(16))) __bf16 v16bf;
typedef __attribute__((ext_vector_type(8)))  __bf16 v8bf;
typedef __attribute__((ext_vector_type(4)))  __bf16 v4bf;
typedef __attribute__((ext_vector_type(8)))  float  v8f;
typedef __attribute__((ext_vector_type(4)))  float  v4f;

static __device__ inline v8f wmma_bf16(v16bf a, v16bf b, v8f c) {
    // (neg_a, A, neg_b, B, c_mod, C, reuse_a, reuse_b)
    return __builtin_amdgcn_wmma_f32_16x16x32_bf16(false, a, false, b, (short)0, c,
                                                   false, false);
}

// Build a 16-element bf16 fragment from two contiguous 8-element (16 B) runs.
static __device__ inline v16bf frag16(const bf16* p0, const bf16* p1) {
    v8bf lo = *(const v8bf*)p0;
    v8bf hi = *(const v8bf*)p1;
    return __builtin_shufflevector(lo, hi, 0, 1, 2, 3, 4, 5, 6, 7,
                                           8, 9, 10, 11, 12, 13, 14, 15);
}

// Raw hardware exp2 (single v_exp_f32, no libm range fixup). Arguments in the
// flash softmax are <= 0 (or -inf on masked lanes), both handled natively.
#if __has_builtin(__builtin_amdgcn_exp2f)
#define FAST_EXP2(x) __builtin_amdgcn_exp2f(x)
#define QSCALE 0.1803368801111204f   // d^-0.5 * log2(e): scores in log2 units
#else
#define FAST_EXP2(x) __expf((x) * 0.6931471805599453f)
#define QSCALE 0.1803368801111204f
#endif

// ---------------------------------------------------------------------------
// Kernel 1: qkv = x @ w_attn + b_attn, fused with
//   q *= d^-0.5*log2e ; k += rel_s[t] ; v stored TRANSPOSED [B*H][64][T].
// Block tile 128(M) x 64(N). Because 768 = 12*64, each block's 64-wide column
// range lies entirely in one region (q/k/v) and one head:
//   region = blockIdx.x/12, h = blockIdx.x%12  -> scalar branch, no per-lane
//   integer division anywhere.
// q/k epilogues repack through LDS so all global stores are b128.
// ---------------------------------------------------------------------------
#define XLD  40
#define WLD  40
#define EPLD 72
#define SMEM_ELEMS 9216   // max(128*XLD + 64*WLD, 128*EPLD) = max(7680, 9216)

__global__ void __launch_bounds__(256)
qkv_kernel(const float* __restrict__ x, const float* __restrict__ w_attn,
           const float* __restrict__ b_attn, const float* __restrict__ rel_emb,
           bf16* __restrict__ qbuf, bf16* __restrict__ kbuf, bf16* __restrict__ vbufT)
{
    __shared__ __align__(16) bf16 SMEM[SMEM_ELEMS];
    bf16* Xs = SMEM;              // 128 x 32 (stride 40), row-major
    bf16* Wt = SMEM + 128 * XLD;  // 64 x 32 (stride 40), W transposed [n][k]

    const int tid  = threadIdx.x;
    const int lane = tid & 31;
    const int wave = tid >> 5;
    const int m0 = blockIdx.y * 128;
    const int n0 = blockIdx.x * 64;
    const int wm = (wave & 3) * 32;   // 4 waves along M
    const int wn = (wave >> 2) * 32;  // 2 waves along N
    const int lc = lane & 15, half = lane >> 4;

    v8f acc[2][2];
    #pragma unroll
    for (int i = 0; i < 2; ++i)
        #pragma unroll
        for (int j = 0; j < 2; ++j) acc[i][j] = (v8f){};

    for (int k0 = 0; k0 < CDIM; k0 += 32) {
        // stage X tile 128x32: float4 global loads -> packed bf16x4 LDS stores
        #pragma unroll
        for (int i = 0; i < 4; ++i) {
            int g = tid + i * 256;            // 0..1023
            int r = g >> 3, c4 = (g & 7) * 4;
            v4f xv = *(const v4f*)&x[(size_t)(m0 + r) * CDIM + k0 + c4];
            *(v4bf*)&Xs[r * XLD + c4] = __builtin_convertvector(xv, v4bf);
        }
        // stage W tile 32x64 TRANSPOSED: column reads (coalesced across lanes)
        #pragma unroll
        for (int i = 0; i < 2; ++i) {
            int g = tid + i * 256;            // 0..511
            int c = g & 63, r4 = (g >> 6) * 4;
            v4f wv;
            #pragma unroll
            for (int j = 0; j < 4; ++j)
                wv[j] = w_attn[(size_t)(k0 + r4 + j) * (3 * CDIM) + n0 + c];
            *(v4bf*)&Wt[c * WLD + r4] = __builtin_convertvector(wv, v4bf);
        }
        __syncthreads();

        v16bf afrag[2], bfrag[2];
        #pragma unroll
        for (int mi = 0; mi < 2; ++mi) {
            const bf16* ap = &Xs[(wm + mi * 16 + lc) * XLD + half * 8];
            afrag[mi] = frag16(ap, ap + 16);
        }
        #pragma unroll
        for (int ni = 0; ni < 2; ++ni) {
            const bf16* bp = &Wt[(wn + ni * 16 + lc) * WLD + half * 8];
            bfrag[ni] = frag16(bp, bp + 16);
        }
        #pragma unroll
        for (int mi = 0; mi < 2; ++mi)
            #pragma unroll
            for (int ni = 0; ni < 2; ++ni)
                acc[mi][ni] = wmma_bf16(afrag[mi], bfrag[ni], acc[mi][ni]);
        __syncthreads();
    }

    // ---- epilogue: region/head are BLOCK-uniform (scalar) ----
    const int region = blockIdx.x / NHEAD;   // 0=q 1=k 2=v
    const int h      = blockIdx.x % NHEAD;

    if (region == 2) {
        // v: direct packed stores into transposed [B*H][64][T] layout
        #pragma unroll
        for (int mi = 0; mi < 2; ++mi) {
            int gmbU = m0 + wm + mi * 16;          // uniform per wave
            int bidx = gmbU >> 10;
            int tb = (gmbU & 1023) + half * 8;
            #pragma unroll
            for (int ni = 0; ni < 2; ++ni) {
                int dd = wn + ni * 16 + lc;        // no division: n0 % 64 == 0
                float bias = b_attn[2 * CDIM + h * DHEAD + dd];
                v8f tv = acc[mi][ni] + bias;
                *(v8bf*)&vbufT[((size_t)((bidx * NHEAD + h) * DHEAD + dd)) * TDIM + tb] =
                    __builtin_convertvector(tv, v8bf);
            }
        }
    } else {
        // q/k: scatter to LDS (bf16), then b128 row stores; k adds the shifted
        // rel_emb at the repack stage with contiguous v4f loads.
        bf16* Es = SMEM;   // 128 x 64, stride EPLD (staging buffers are dead)
        #pragma unroll
        for (int mi = 0; mi < 2; ++mi) {
            int rowb = wm + mi * 16 + half * 8;
            #pragma unroll
            for (int ni = 0; ni < 2; ++ni) {
                int dd = wn + ni * 16 + lc;
                float bias = b_attn[region * CDIM + h * DHEAD + dd];
                v8f tv = acc[mi][ni] + bias;
                if (region == 0) tv = tv * QSCALE;   // d^-0.5 * log2e
                #pragma unroll
                for (int r = 0; r < 8; ++r)
                    Es[(rowb + r) * EPLD + dd] = (bf16)tv[r];
            }
        }
        __syncthreads();
        int row = tid >> 1, cb = (tid & 1) * 32;     // 256 threads: 128 rows x 2
        int gm = m0 + row;
        int bidx = gm >> 10, tt = gm & 1023;
        const bf16* src = Es + row * EPLD + cb;
        bf16* dstb = (region == 0 ? qbuf : kbuf) +
                     ((size_t)(bidx * NHEAD + h) * TDIM + tt) * DHEAD + cb;
        if (region == 0) {
            #pragma unroll
            for (int u = 0; u < 4; ++u)
                *(v8bf*)(dstb + u * 8) = *(const v8bf*)(src + u * 8);
        } else {
            int tr = (tt > 0) ? (tt - 1) : 0;        // clamp; masked below
            const float* relp = rel_emb + (size_t)tr * DHEAD + cb;
            float wgt = (tt > 0) ? 1.0f : 0.0f;
            #pragma unroll
            for (int u = 0; u < 4; ++u) {
                v8f f = __builtin_convertvector(*(const v8bf*)(src + u * 8), v8f);
                v4f r0 = *(const v4f*)(relp + u * 8);
                v4f r1 = *(const v4f*)(relp + u * 8 + 4);
                v8f rel8 = __builtin_shufflevector(r0, r1, 0, 1, 2, 3, 4, 5, 6, 7);
                f = f + rel8 * wgt;
                *(v8bf*)(dstb + u * 8) = __builtin_convertvector(f, v8bf);
            }
        }
    }
}

// ---------------------------------------------------------------------------
// Kernel 2: causal flash attention, y = softmax(q k'^T) v  per (b,h).
// 4 waves/block, 16 query rows/wave, 32 keys/step. Q/K fragments are direct
// global b128 loads (row-major [t][64]); V fragments are direct global b128
// loads from the transposed [64][T] buffer. Only P round-trips LDS
// (C-layout -> A-layout), ordered by s_wait_dscnt. Softmax runs in log2
// units (q pre-scaled by log2e) so every exp is one raw v_exp_f32.
// ---------------------------------------------------------------------------
#define PLD 72

__global__ void __launch_bounds__(128)
attn_kernel(const bf16* __restrict__ qbuf, const bf16* __restrict__ kbuf,
            const bf16* __restrict__ vbufT, bf16* __restrict__ ybuf)
{
    __shared__ __align__(16) bf16 Plds[4][16 * PLD];

    const int tid  = threadIdx.x;
    const int lane = tid & 31;
    const int wave = tid >> 5;
    const int bh = blockIdx.y;                       // 0..95
    const int i0 = blockIdx.x * 64 + wave * 16;      // first query row of wave
    const int lc = lane & 15, half = lane >> 4;

    const bf16* kbase = kbuf  + (size_t)bh * TDIM * DHEAD;
    const bf16* vbase = vbufT + (size_t)bh * DHEAD * TDIM;
    bf16* pl = Plds[wave];

    // Q A-fragments for rows i0..i0+15 (K-dim 64 -> 2 fragments), 2xb128 each
    const bf16* qrow = qbuf + ((size_t)bh * TDIM + i0 + lc) * DHEAD;
    v16bf qa0 = frag16(qrow + half * 8,      qrow + 16 + half * 8);
    v16bf qa1 = frag16(qrow + 32 + half * 8, qrow + 48 + half * 8);

    v8f accY[4];
    #pragma unroll
    for (int t = 0; t < 4; ++t) accY[t] = (v8f){};
    float m_i[8], l_i[8];
    #pragma unroll
    for (int r = 0; r < 8; ++r) { m_i[r] = -__builtin_inff(); l_i[r] = 0.0f; }

    for (int j0 = 0; j0 < i0 + 16; j0 += 32) {
        if (j0 + 32 < i0 + 16) { // prefetch next k/v tiles (global_prefetch_b8)
            __builtin_prefetch((const void*)(kbase + (size_t)(j0 + 32 + lane) * DHEAD), 0, 1);
            __builtin_prefetch((const void*)(vbase + (size_t)(lane * 2) * TDIM + j0 + 32), 0, 1);
        }
        // S = q @ k'^T for 32 keys -> two 16x16 tiles (scores in log2 units)
        v8f accS[2] = { (v8f){}, (v8f){} };
        #pragma unroll
        for (int nt = 0; nt < 2; ++nt) {
            const bf16* kp = kbase + (size_t)(j0 + nt * 16 + lc) * DHEAD + half * 8;
            accS[nt] = wmma_bf16(qa0, frag16(kp, kp + 16),      accS[nt]);
            accS[nt] = wmma_bf16(qa1, frag16(kp + 32, kp + 48), accS[nt]);
        }
        // online softmax, per row (elem r, 16-lane half); exp2 = v_exp_f32
        #pragma unroll
        for (int r = 0; r < 8; ++r) {
            int irow = i0 + half * 8 + r;
            float s0 = accS[0][r], s1 = accS[1][r];
            if (j0 + lc > irow)      s0 = -__builtin_inff();
            if (j0 + 16 + lc > irow) s1 = -__builtin_inff();
            float mx = fmaxf(s0, s1);
            mx = fmaxf(mx, __shfl_xor(mx, 1));
            mx = fmaxf(mx, __shfl_xor(mx, 2));
            mx = fmaxf(mx, __shfl_xor(mx, 4));
            mx = fmaxf(mx, __shfl_xor(mx, 8));
            float mnew = fmaxf(m_i[r], mx);
            float cf = FAST_EXP2(m_i[r] - mnew);
            float p0 = FAST_EXP2(s0 - mnew);
            float p1 = FAST_EXP2(s1 - mnew);
            float ps = p0 + p1;
            ps += __shfl_xor(ps, 1);
            ps += __shfl_xor(ps, 2);
            ps += __shfl_xor(ps, 4);
            ps += __shfl_xor(ps, 8);
            l_i[r] = l_i[r] * cf + ps;
            m_i[r] = mnew;
            #pragma unroll
            for (int t = 0; t < 4; ++t) accY[t][r] *= cf;
            int prow = half * 8 + r;     // C layout -> LDS row-major P[16][32]
            pl[prow * PLD + lc]      = (bf16)p0;
            pl[prow * PLD + 16 + lc] = (bf16)p1;
        }
        asm volatile("s_wait_dscnt 0" ::: "memory");
        // P (16x32) as A fragment from LDS: 2 x ds_load_b128
        const bf16* pp = pl + lc * PLD + half * 8;
        v16bf pa = frag16(pp, pp + 16);
        // Y += P @ V : V fragments contiguous in transposed layout, 2xb128 each
        #pragma unroll
        for (int t = 0; t < 4; ++t) {
            const bf16* vp = vbase + (size_t)(t * 16 + lc) * TDIM + j0 + half * 8;
            accY[t] = wmma_bf16(pa, frag16(vp, vp + 16), accY[t]);
        }
    }

    // epilogue: normalize, repack through LDS so global y stores are b128
    int b = bh / NHEAD, h = bh % NHEAD;
    #pragma unroll
    for (int r = 0; r < 8; ++r) {
        float inv = 1.0f / l_i[r];
        int prow = half * 8 + r;
        #pragma unroll
        for (int t = 0; t < 4; ++t)
            pl[prow * PLD + t * 16 + lc] = (bf16)(accY[t][r] * inv);
    }
    asm volatile("s_wait_dscnt 0" ::: "memory");
    {
        int row = lane >> 1, cb = (lane & 1) * 32;   // 32 lanes cover 16 rows x 64
        const bf16* src = pl + row * PLD + cb;
        v8bf y0 = *(const v8bf*)(src);
        v8bf y1 = *(const v8bf*)(src + 8);
        v8bf y2 = *(const v8bf*)(src + 16);
        v8bf y3 = *(const v8bf*)(src + 24);
        bf16* dst = ybuf + ((size_t)(b * TDIM + i0 + row)) * CDIM + h * DHEAD + cb;
        *(v8bf*)(dst)      = y0;
        *(v8bf*)(dst + 8)  = y1;
        *(v8bf*)(dst + 16) = y2;
        *(v8bf*)(dst + 24) = y3;
    }
}

// ---------------------------------------------------------------------------
// Kernel 3: out = y @ w_proj + b_proj   (8192 x 768 x 768), fp32 output.
// Same tiling as kernel 1; A operand already bf16. Output stores are
// lane-coalesced (16 consecutive floats per half-wave).
// ---------------------------------------------------------------------------
__global__ void __launch_bounds__(256)
proj_kernel(const bf16* __restrict__ ybuf, const float* __restrict__ w_proj,
            const float* __restrict__ b_proj, float* __restrict__ out)
{
    __shared__ __align__(16) bf16 Ys[128 * XLD];
    __shared__ __align__(16) bf16 Wt[64 * WLD];

    const int tid  = threadIdx.x;
    const int lane = tid & 31;
    const int wave = tid >> 5;
    const int m0 = blockIdx.y * 128;
    const int n0 = blockIdx.x * 64;
    const int wm = (wave & 3) * 32;
    const int wn = (wave >> 2) * 32;
    const int lc = lane & 15, half = lane >> 4;

    v8f acc[2][2];
    #pragma unroll
    for (int i = 0; i < 2; ++i)
        #pragma unroll
        for (int j = 0; j < 2; ++j) acc[i][j] = (v8f){};

    for (int k0 = 0; k0 < CDIM; k0 += 32) {
        #pragma unroll
        for (int i = 0; i < 4; ++i) {
            int g = tid + i * 256;
            int r = g >> 3, c4 = (g & 7) * 4;
            *(v4bf*)&Ys[r * XLD + c4] =
                *(const v4bf*)&ybuf[(size_t)(m0 + r) * CDIM + k0 + c4];
        }
        #pragma unroll
        for (int i = 0; i < 2; ++i) {
            int g = tid + i * 256;
            int c = g & 63, r4 = (g >> 6) * 4;
            v4f wv;
            #pragma unroll
            for (int j = 0; j < 4; ++j)
                wv[j] = w_proj[(size_t)(k0 + r4 + j) * CDIM + n0 + c];
            *(v4bf*)&Wt[c * WLD + r4] = __builtin_convertvector(wv, v4bf);
        }
        __syncthreads();

        v16bf afrag[2], bfrag[2];
        #pragma unroll
        for (int mi = 0; mi < 2; ++mi) {
            const bf16* ap = &Ys[(wm + mi * 16 + lc) * XLD + half * 8];
            afrag[mi] = frag16(ap, ap + 16);
        }
        #pragma unroll
        for (int ni = 0; ni < 2; ++ni) {
            const bf16* bp = &Wt[(wn + ni * 16 + lc) * WLD + half * 8];
            bfrag[ni] = frag16(bp, bp + 16);
        }
        #pragma unroll
        for (int mi = 0; mi < 2; ++mi)
            #pragma unroll
            for (int ni = 0; ni < 2; ++ni)
                acc[mi][ni] = wmma_bf16(afrag[mi], bfrag[ni], acc[mi][ni]);
        __syncthreads();
    }

    #pragma unroll
    for (int mi = 0; mi < 2; ++mi) {
        #pragma unroll
        for (int ni = 0; ni < 2; ++ni) {
            int gn = n0 + wn + ni * 16 + lc;
            float bias = b_proj[gn];
            #pragma unroll
            for (int r = 0; r < 8; ++r) {
                int gm = m0 + wm + mi * 16 + half * 8 + r;
                out[(size_t)gm * CDIM + gn] = acc[mi][ni][r] + bias;
            }
        }
    }
}

// ---------------------------------------------------------------------------
extern "C" void kernel_launch(void* const* d_in, const int* in_sizes, int n_in,
                              void* d_out, int out_size, void* d_ws, size_t ws_size,
                              hipStream_t stream) {
    const float* x       = (const float*)d_in[0];
    const float* w_attn  = (const float*)d_in[1];
    const float* b_attn  = (const float*)d_in[2];
    const float* w_proj  = (const float*)d_in[3];
    const float* b_proj  = (const float*)d_in[4];
    const float* rel_emb = (const float*)d_in[5];
    float* out = (float*)d_out;

    const size_t QKV_ELEMS = (size_t)8 * NHEAD * TDIM * DHEAD; // 6,291,456
    bf16* qbuf  = (bf16*)d_ws;
    bf16* kbuf  = qbuf + QKV_ELEMS;
    bf16* vbufT = kbuf + QKV_ELEMS;  // transposed: [B*H][64][T]
    bf16* ybuf  = vbufT + QKV_ELEMS; // 8192*768 elements == QKV_ELEMS

    // qkv: M=8192 (64 tiles of 128), N=2304 (36 tiles of 64)
    qkv_kernel<<<dim3(36, 64), 256, 0, stream>>>(x, w_attn, b_attn, rel_emb,
                                                 qbuf, kbuf, vbufT);
    // attention: 16 row-blocks of 64 x 96 (b,h)
    attn_kernel<<<dim3(16, 96), 128, 0, stream>>>(qbuf, kbuf, vbufT, ybuf);
    // out projection: N=768 (12 tiles of 64), M=8192 (64 tiles of 128)
    proj_kernel<<<dim3(12, 64), 256, 0, stream>>>(ybuf, w_proj, b_proj, out);
}
